// DiTSiToBlock_52278341927304
// MI455X (gfx1250) — matrix operand order repaired and checked by reading, not compiled, not run on hardware
//
#include <hip/hip_runtime.h>
#include <hip/hip_bf16.h>
#include <math.h>

#define DEV __device__ __forceinline__

typedef _Float16 v8h  __attribute__((ext_vector_type(8)));
typedef _Float16 v16h __attribute__((ext_vector_type(16)));
typedef float    v8f  __attribute__((ext_vector_type(8)));

constexpr int B_    = 16;
constexpr int N_    = 1024;
constexpr int D_    = 1152;
constexpr int H_    = 16;
constexpr int DH_   = 72;
constexpr int DHP_  = 96;   // K-padded head dim for q/k (3 x 32)
constexpr int DVP_  = 80;   // N-padded head dim for v^T rows (5 x 16)
constexpr int GW_   = 32;
constexpr int NW_   = 256;  // windows
constexpr int WS_   = 4;
constexpr int NS_   = 768;  // src tokens
constexpr int T_    = 512;  // kept tokens (NW + 256)
constexpr int PR_   = 512;  // pruned
constexpr int KEEP_ = 256;  // src kept
constexpr float ALPHA_ = 0.1f;
constexpr float BETA_  = 1.0f;
constexpr int SROW_ = 516;  // LDS score row stride (floats), conflict-avoiding

// ---------------- WMMA helpers (CDNA5 wave32 layouts) ----------------

DEV v8f wmma_f16(v16h a, v16h b, v8f c) {
  return __builtin_amdgcn_wmma_f32_16x16x32_f16(false, a, false, b, (short)0, c, false, false);
}

// A fragment 16x32 f16: lane L holds row M=L%16; lanes<16: K 0..7 & 16..23, lanes>=16: K 8..15 & 24..31
DEV v16h load_a16(const _Float16* rowBase, int k0, int lane) {
  int ko = (lane >> 4) << 3;
  v8h lo = *(const v8h*)(rowBase + k0 + ko);
  v8h hi = *(const v8h*)(rowBase + k0 + ko + 16);
  return __builtin_shufflevector(lo, hi, 0,1,2,3,4,5,6,7,8,9,10,11,12,13,14,15);
}
// B fragment 32x16 f16 from B^T row-major (N,K): lane L holds col N=L%16; lanes<16: K 0..7 & 16..23, lanes>=16: K 8..15 & 24..31
DEV v16h load_b16(const _Float16* nrowBase, int k0, int lane) {
  int kb = (lane >> 4) << 3;
  v8h lo = *(const v8h*)(nrowBase + k0 + kb);
  v8h hi = *(const v8h*)(nrowBase + k0 + kb + 16);
  return __builtin_shufflevector(lo, hi, 0,1,2,3,4,5,6,7,8,9,10,11,12,13,14,15);
}

DEV float gelu_tanh(float x) {
  float x3 = x * x * x;
  return 0.5f * x * (1.f + tanhf(0.7978845608028654f * (x + 0.044715f * x3)));
}

// ---------------- K1: row-normalize x -> f16 ----------------
__global__ __launch_bounds__(128) void k_xnorm(const float* __restrict__ x, _Float16* __restrict__ xn) {
  __shared__ float sbuf[4];
  size_t row = blockIdx.x;                    // B_*N_
  const float* xr = x + row * D_;
  float ss = 0.f;
  for (int d = threadIdx.x; d < D_; d += 128) { float v = xr[d]; ss += v * v; }
  for (int o = 16; o > 0; o >>= 1) ss += __shfl_xor(ss, o, 32);
  int lane = threadIdx.x & 31, w = threadIdx.x >> 5;
  if (lane == 0) sbuf[w] = ss;
  __syncthreads();
  float inv = 1.f / (sqrtf(sbuf[0] + sbuf[1] + sbuf[2] + sbuf[3]) + 1e-6f);
  _Float16* orow = xn + row * D_;
  for (int d = threadIdx.x; d < D_; d += 128) orow[d] = (_Float16)(xr[d] * inv);
}

// ---------------- K2: window scores -> dst/src indices ----------------
__global__ __launch_bounds__(128) void k_window(const _Float16* __restrict__ xn, const float* __restrict__ noise,
                                                int* __restrict__ dst_idx, int* __restrict__ src_idx) {
  __shared__ float sred[4][4];
  int id = blockIdx.x;                        // B_*NW_
  int b = id / NW_, wdw = id % NW_;
  int wy = wdw / 16, wx = wdw % 16;
  int tok[4];
  tok[0] = (wy*2+0)*GW_ + wx*2+0;
  tok[1] = (wy*2+0)*GW_ + wx*2+1;
  tok[2] = (wy*2+1)*GW_ + wx*2+0;
  tok[3] = (wy*2+1)*GW_ + wx*2+1;
  const _Float16* base = xn + (size_t)b * N_ * D_;
  float acc[4] = {0.f, 0.f, 0.f, 0.f};
  for (int d = threadIdx.x; d < D_; d += 128) {
    float v0 = (float)base[(size_t)tok[0]*D_ + d];
    float v1 = (float)base[(size_t)tok[1]*D_ + d];
    float v2 = (float)base[(size_t)tok[2]*D_ + d];
    float v3 = (float)base[(size_t)tok[3]*D_ + d];
    float s = v0 + v1 + v2 + v3;
    acc[0] += v0 * s; acc[1] += v1 * s; acc[2] += v2 * s; acc[3] += v3 * s;
  }
  #pragma unroll
  for (int i = 0; i < 4; i++)
    for (int o = 16; o > 0; o >>= 1) acc[i] += __shfl_xor(acc[i], o, 32);
  int lane = threadIdx.x & 31, w = threadIdx.x >> 5;
  if (lane == 0) { sred[w][0]=acc[0]; sred[w][1]=acc[1]; sred[w][2]=acc[2]; sred[w][3]=acc[3]; }
  __syncthreads();
  if (threadIdx.x == 0) {
    int bestL = 0; float bestS = -1e30f;
    for (int i = 0; i < 4; i++) {
      float dot = (sred[0][i] + sred[1][i] + sred[2][i] + sred[3][i]) * 0.25f;
      float sc = BETA_ * dot + ALPHA_ * noise[(size_t)(b*NW_ + wdw)*WS_ + i];
      if (sc > bestS) { bestS = sc; bestL = i; }
    }
    dst_idx[b*NW_ + wdw] = tok[bestL];
    int j = 0;
    for (int i = 0; i < 4; i++) if (i != bestL) { src_idx[b*NS_ + wdw*3 + j] = tok[i]; j++; }
  }
}

// ---------------- K3: similarity GEMM (gathered-row WMMA) ----------------
__global__ __launch_bounds__(256) void k_sim(const _Float16* __restrict__ xn, const int* __restrict__ src_idx,
                                             const int* __restrict__ dst_idx, float* __restrict__ sim) {
  int lane = threadIdx.x & 31, w = threadIdx.x >> 5;
  int g = blockIdx.x * 8 + w;                 // B_*768 tiles (48 M x 16 N per batch)
  int b = g / 768, t = g % 768;
  int tm = t >> 4, tn = t & 15;
  int srow = src_idx[b*NS_ + tm*16 + (lane & 15)];
  int drow = dst_idx[b*NW_ + tn*16 + (lane & 15)];
  const _Float16* A  = xn + ((size_t)b*N_ + srow) * D_;
  const _Float16* Bp = xn + ((size_t)b*N_ + drow) * D_;
  v8f acc = {};
  for (int k0 = 0; k0 < D_; k0 += 32)
    acc = wmma_f16(load_a16(A, k0, lane), load_b16(Bp, k0, lane), acc);
  int col  = tn*16 + (lane & 15);
  int mrow = tm*16 + ((lane >> 4) << 3);
  float* out = sim + (size_t)b * NS_ * NW_;
  #pragma unroll
  for (int r = 0; r < 8; r++) out[(size_t)(mrow + r)*NW_ + col] = acc[r];
}

// ---------------- K4: row max + argmax over 256 ----------------
__global__ __launch_bounds__(256) void k_rowmax(const float* __restrict__ sim, float* __restrict__ maxsim,
                                                int* __restrict__ best) {
  int lane = threadIdx.x & 31, w = threadIdx.x >> 5;
  int row = blockIdx.x * 8 + w;               // B_*768
  const float* r = sim + (size_t)row * NW_;
  float mv = -1e30f; int mi = 0;
  for (int c = lane; c < NW_; c += 32) { float v = r[c]; if (v > mv) { mv = v; mi = c; } }
  for (int o = 16; o > 0; o >>= 1) {
    float ov = __shfl_xor(mv, o, 32); int oi = __shfl_xor(mi, o, 32);
    if (ov > mv || (ov == mv && oi < mi)) { mv = ov; mi = oi; }
  }
  if (lane == 0) { maxsim[row] = mv; best[row] = mi; }
}

// ---------------- K5: per-batch bitonic sort (desc, idx tie-break) ----------------
__global__ __launch_bounds__(512) void k_sort(const float* __restrict__ maxsim, const int* __restrict__ best,
                                              const int* __restrict__ src_idx, int* __restrict__ src_kept,
                                              int* __restrict__ src_pruned, int* __restrict__ pruned_best) {
  __shared__ float key[1024];
  __shared__ int   val[1024];
  int b = blockIdx.x;
  for (int s = 0; s < 2; s++) {
    int i = threadIdx.x + s * 512;
    if (i < NS_) { key[i] = maxsim[b*NS_ + i]; val[i] = i; }
    else         { key[i] = -3e38f;            val[i] = i; }
  }
  __syncthreads();
  for (int k = 2; k <= 1024; k <<= 1) {
    for (int j = k >> 1; j > 0; j >>= 1) {
      for (int s = 0; s < 2; s++) {
        int i = threadIdx.x + s * 512;
        int ixj = i ^ j;
        if (ixj > i) {
          bool up = ((i & k) == 0);
          float ka = key[i], kb = key[ixj]; int va = val[i], vb = val[ixj];
          bool bfirst = (kb > ka) || (kb == ka && vb < va);   // elem[ixj] precedes elem[i] in desired order
          if (up == bfirst) { key[i] = kb; key[ixj] = ka; val[i] = vb; val[ixj] = va; }
        }
      }
      __syncthreads();
    }
  }
  int t = threadIdx.x;
  { int o = val[t]; src_pruned[b*PR_ + t] = src_idx[b*NS_ + o]; pruned_best[b*PR_ + t] = best[b*NS_ + o]; }
  if (t < KEEP_) { int o = val[PR_ + t]; src_kept[b*KEEP_ + t] = src_idx[b*NS_ + o]; }
}

// ---------------- K6: gather kept tokens ----------------
__global__ __launch_bounds__(128) void k_gather(const float* __restrict__ x, const int* __restrict__ dst_idx,
                                                const int* __restrict__ src_kept, float* __restrict__ xk) {
  int id = blockIdx.x; int b = id / T_, i = id % T_;
  int tok = (i < NW_) ? dst_idx[b*NW_ + i] : src_kept[b*KEEP_ + (i - NW_)];
  const float* src = x  + ((size_t)b*N_ + tok) * D_;
  float*       dst = xk + ((size_t)b*T_ + i)   * D_;
  for (int d = threadIdx.x; d < D_; d += 128) dst[d] = src[d];
}

// ---------------- K7: LayerNorm (f32 in -> f16 out) ----------------
__global__ __launch_bounds__(128) void k_ln(const float* __restrict__ X, const float* __restrict__ g,
                                            const float* __restrict__ bt, _Float16* __restrict__ Y) {
  __shared__ float s1[4], s2[4];
  size_t row = blockIdx.x;
  const float* xr = X + row * D_;
  float sm = 0.f, sq = 0.f;
  for (int d = threadIdx.x; d < D_; d += 128) { float v = xr[d]; sm += v; sq += v * v; }
  for (int o = 16; o > 0; o >>= 1) { sm += __shfl_xor(sm, o, 32); sq += __shfl_xor(sq, o, 32); }
  int lane = threadIdx.x & 31, w = threadIdx.x >> 5;
  if (lane == 0) { s1[w] = sm; s2[w] = sq; }
  __syncthreads();
  float mean = (s1[0]+s1[1]+s1[2]+s1[3]) * (1.f / D_);
  float var  = (s2[0]+s2[1]+s2[2]+s2[3]) * (1.f / D_) - mean * mean;
  float rs = rsqrtf(var + 1e-6f);
  _Float16* yr = Y + row * D_;
  for (int d = threadIdx.x; d < D_; d += 128) yr[d] = (_Float16)(((xr[d] - mean) * rs) * g[d] + bt[d]);
}

// ---------------- K8: weight transpose + f32->f16 ----------------
__global__ __launch_bounds__(256) void k_transpose(const float* __restrict__ w, _Float16* __restrict__ wt,
                                                   int K, int Nn) {
  __shared__ float tile[32][33];
  int kx = blockIdx.x * 32, nx = blockIdx.y * 32;
  int tx = threadIdx.x & 31, ty = threadIdx.x >> 5;   // 32 x 8
  for (int i = ty; i < 32; i += 8) tile[i][tx] = w[(size_t)(kx + i) * Nn + nx + tx];
  __syncthreads();
  for (int i = ty; i < 32; i += 8) wt[(size_t)(nx + i) * K + kx + tx] = (_Float16)tile[tx][i];
}

// ---------------- K9: WMMA GEMM, 128x64 block tile, 8 waves, fused epilogue ----------------
// mode 0: out f16 (+bias); mode 1: gelu -> f16 (+bias); mode 2: +bias +res -> f32
__global__ __launch_bounds__(256) void k_gemm(const _Float16* __restrict__ A, const _Float16* __restrict__ Bt,
                                              const float* __restrict__ bias, const float* __restrict__ res,
                                              _Float16* __restrict__ outH, float* __restrict__ outF,
                                              int M, int Nn, int K, int mode) {
  int lane = threadIdx.x & 31, w = threadIdx.x >> 5;
  int wm = w >> 1, wn = w & 1;
  int m_base = blockIdx.x * 128 + wm * 32;
  int n_base = blockIdx.y * 64  + wn * 32;
  const _Float16* a0 = A  + (size_t)(m_base + (lane & 15)) * K;
  const _Float16* a1 = a0 + (size_t)16 * K;
  const _Float16* b0 = Bt + (size_t)(n_base + (lane & 15)) * K;
  const _Float16* b1 = b0 + (size_t)16 * K;
  v8f acc00 = {}, acc01 = {}, acc10 = {}, acc11 = {};
  for (int k0 = 0; k0 < K; k0 += 32) {
    __builtin_prefetch(a0 + k0 + 256, 0, 0);
    __builtin_prefetch(b0 + k0 + 256, 0, 0);
    v16h af0 = load_a16(a0, k0, lane);
    v16h af1 = load_a16(a1, k0, lane);
    v16h bf0 = load_b16(b0, k0, lane);
    v16h bf1 = load_b16(b1, k0, lane);
    acc00 = wmma_f16(af0, bf0, acc00);
    acc01 = wmma_f16(af0, bf1, acc01);
    acc10 = wmma_f16(af1, bf0, acc10);
    acc11 = wmma_f16(af1, bf1, acc11);
  }
  int coln = n_base + (lane & 15);
  int mro  = (lane >> 4) << 3;
  v8f accs[2][2] = {{acc00, acc01}, {acc10, acc11}};
  #pragma unroll
  for (int mi = 0; mi < 2; mi++) {
    #pragma unroll
    for (int ni = 0; ni < 2; ni++) {
      v8f acc = accs[mi][ni];
      int n = coln + ni * 16;
      float bv = bias ? bias[n] : 0.f;
      #pragma unroll
      for (int r = 0; r < 8; r++) {
        int m = m_base + mi * 16 + mro + r;
        float v = acc[r] + bv;
        size_t idx = (size_t)m * Nn + n;
        if (mode == 0)      outH[idx] = (_Float16)v;
        else if (mode == 1) outH[idx] = (_Float16)gelu_tanh(v);
        else                outF[idx] = v + res[idx];
      }
    }
  }
}

// ---------------- K10: repack qkv -> padded per-head q/k and transposed v ----------------
__global__ __launch_bounds__(128) void k_repack(const _Float16* __restrict__ qkv, _Float16* __restrict__ qp,
                                                _Float16* __restrict__ kp, _Float16* __restrict__ vt) {
  int id = blockIdx.x;                        // b*H_*T_ + h*T_ + t
  int t = id % T_; int bh = id / T_;
  int b = bh / H_, h = bh % H_;
  const _Float16* row = qkv + (size_t)(b * T_ + t) * (3 * D_);
  int d = threadIdx.x;
  if (d < DHP_) {
    _Float16 qv = (d < DH_) ? row[h*DH_ + d]       : (_Float16)0.f;
    _Float16 kv = (d < DH_) ? row[D_ + h*DH_ + d]  : (_Float16)0.f;
    qp[((size_t)bh*T_ + t)*DHP_ + d] = qv;
    kp[((size_t)bh*T_ + t)*DHP_ + d] = kv;
  }
  if (d < DVP_) {
    _Float16 vv = (d < DH_) ? row[2*D_ + h*DH_ + d] : (_Float16)0.f;
    vt[((size_t)bh*DVP_ + d)*T_ + t] = vv;
  }
}

// ---------------- K11: attention: S=QK^T (WMMA) -> softmax (LDS) -> O=PV (WMMA) ----------------
__global__ __launch_bounds__(32) void k_attn(const _Float16* __restrict__ qp, const _Float16* __restrict__ kp,
                                             const _Float16* __restrict__ vt, _Float16* __restrict__ o) {
  __shared__ float s[16 * SROW_];             // 33KB: 16 score rows of 512 (+pad)
  int bh = blockIdx.x; int m0 = blockIdx.y * 16;
  int lane = threadIdx.x;
  const _Float16* qrow = qp + ((size_t)bh*T_ + m0 + (lane & 15)) * DHP_;
  v16h qf0 = load_a16(qrow, 0,  lane);
  v16h qf1 = load_a16(qrow, 32, lane);
  v16h qf2 = load_a16(qrow, 64, lane);
  const float scale = 0.11785113019775793f;   // 1/sqrt(72)
  for (int j = 0; j < T_ / 16; j++) {
    const _Float16* krow = kp + ((size_t)bh*T_ + j*16 + (lane & 15)) * DHP_;
    v8f acc = {};
    acc = wmma_f16(qf0, load_b16(krow, 0,  lane), acc);
    acc = wmma_f16(qf1, load_b16(krow, 32, lane), acc);
    acc = wmma_f16(qf2, load_b16(krow, 64, lane), acc);
    int mr = (lane >> 4) << 3;
    #pragma unroll
    for (int r = 0; r < 8; r++) s[(mr + r)*SROW_ + j*16 + (lane & 15)] = acc[r] * scale;
  }
  __syncthreads();
  if (lane < 16) {                            // in-place f32 -> f16 softmax per row
    float* row = s + lane * SROW_;
    _Float16* rowh = (_Float16*)row;
    float mx = -1e30f;
    for (int t = 0; t < T_; t++) mx = fmaxf(mx, row[t]);
    float sum = 0.f;
    for (int t = 0; t < T_; t++) { float e = __expf(row[t] - mx); sum += e; rowh[t] = (_Float16)e; }
    float inv = 1.f / sum;
    for (int t = 0; t < T_; t++) rowh[t] = (_Float16)((float)rowh[t] * inv);
  }
  __syncthreads();
  int b = bh / H_, h = bh % H_;
  const _Float16* prow = (const _Float16*)(s + (size_t)(lane & 15) * SROW_);
  for (int nt = 0; nt < 5; nt++) {
    const _Float16* vrow = vt + ((size_t)bh*DVP_ + nt*16 + (lane & 15)) * T_;
    v8f acc = {};
    for (int k0 = 0; k0 < T_; k0 += 32) {
      int ko = (lane >> 4) << 3;
      v8h lo = *(const v8h*)(prow + k0 + ko);
      v8h hi = *(const v8h*)(prow + k0 + ko + 16);
      v16h pa = __builtin_shufflevector(lo, hi, 0,1,2,3,4,5,6,7,8,9,10,11,12,13,14,15);
      acc = wmma_f16(pa, load_b16(vrow, k0, lane), acc);
    }
    int n = nt*16 + (lane & 15);
    if (n < DH_) {
      int mr = (lane >> 4) << 3;
      #pragma unroll
      for (int r = 0; r < 8; r++)
        o[((size_t)b*T_ + m0 + mr + r)*D_ + h*DH_ + n] = (_Float16)acc[r];
    }
  }
}

// ---------------- K12: scatter back to full token grid ----------------
__global__ __launch_bounds__(128) void k_scatter(const float* __restrict__ xout, const int* __restrict__ dst_idx,
                                                 const int* __restrict__ src_kept, const int* __restrict__ src_pruned,
                                                 const int* __restrict__ pruned_best, float* __restrict__ out) {
  int id = blockIdx.x; int b = id / N_, i = id % N_;
  int orow, srow;
  if (i < NW_)               { orow = dst_idx[b*NW_ + i];                 srow = i; }
  else if (i < NW_ + KEEP_)  { int j = i - NW_; orow = src_kept[b*KEEP_ + j]; srow = NW_ + j; }
  else                       { int p = i - NW_ - KEEP_; orow = src_pruned[b*PR_ + p]; srow = pruned_best[b*PR_ + p]; }
  const float* src = xout + ((size_t)b*T_ + srow) * D_;
  float*       dst = out  + ((size_t)b*N_ + orow) * D_;
  for (int d = threadIdx.x; d < D_; d += 128) dst[d] = src[d];
}

// ---------------- launch ----------------
extern "C" void kernel_launch(void* const* d_in, const int* in_sizes, int n_in,
                              void* d_out, int out_size, void* d_ws, size_t ws_size,
                              hipStream_t stream) {
  (void)in_sizes; (void)n_in; (void)out_size; (void)ws_size;
  const float* x      = (const float*)d_in[0];
  const float* noise  = (const float*)d_in[1];
  const float* ln1_g  = (const float*)d_in[2];
  const float* ln1_b  = (const float*)d_in[3];
  const float* ln2_g  = (const float*)d_in[4];
  const float* ln2_b  = (const float*)d_in[5];
  const float* w_qkv  = (const float*)d_in[6];
  const float* b_qkv  = (const float*)d_in[7];
  const float* w_proj = (const float*)d_in[8];
  const float* b_proj = (const float*)d_in[9];
  const float* w_fc1  = (const float*)d_in[10];
  const float* b_fc1  = (const float*)d_in[11];
  const float* w_fc2  = (const float*)d_in[12];
  const float* b_fc2  = (const float*)d_in[13];
  float* out = (float*)d_out;

  char* base = (char*)d_ws;
  size_t off = 0;
  auto carve = [&](size_t bytes) -> char* {
    char* r = base + off; off += (bytes + 255) & ~(size_t)255; return r;
  };
  _Float16* xn_h   = (_Float16*)carve((size_t)B_*N_*D_*2);
  _Float16* qkv_h  = (_Float16*)carve((size_t)B_*T_*3*D_*2);
  _Float16* y_h    = (_Float16*)carve((size_t)B_*T_*D_*2);
  _Float16* o_h    = (_Float16*)carve((size_t)B_*T_*D_*2);
  _Float16* h1_h   = (_Float16*)carve((size_t)B_*T_*4*D_*2);
  _Float16* qp     = (_Float16*)carve((size_t)B_*H_*T_*DHP_*2);
  _Float16* kp     = (_Float16*)carve((size_t)B_*H_*T_*DHP_*2);
  _Float16* vt     = (_Float16*)carve((size_t)B_*H_*DVP_*T_*2);
  float*    xk     = (float*)carve((size_t)B_*T_*D_*4);
  float*    x1     = (float*)carve((size_t)B_*T_*D_*4);
  float*    xout   = (float*)carve((size_t)B_*T_*D_*4);
  _Float16* wT_qkv  = (_Float16*)carve((size_t)D_*3*D_*2);
  _Float16* wT_proj = (_Float16*)carve((size_t)D_*D_*2);
  _Float16* wT_fc1  = (_Float16*)carve((size_t)D_*4*D_*2);
  _Float16* wT_fc2  = (_Float16*)carve((size_t)4*D_*D_*2);
  float* sim    = (float*)carve((size_t)B_*NS_*NW_*4);
  float* maxsim = (float*)carve((size_t)B_*NS_*4);
  int* best     = (int*)carve((size_t)B_*NS_*4);
  int* dsti     = (int*)carve((size_t)B_*NW_*4);
  int* srci     = (int*)carve((size_t)B_*NS_*4);
  int* skept    = (int*)carve((size_t)B_*KEEP_*4);
  int* sprn     = (int*)carve((size_t)B_*PR_*4);
  int* pbest    = (int*)carve((size_t)B_*PR_*4);

  // Weight transposes (f32 -> f16, N-major so B-fragments are contiguous)
  k_transpose<<<dim3(D_/32, 3*D_/32),  256, 0, stream>>>(w_qkv,  wT_qkv,  D_,   3*D_);
  k_transpose<<<dim3(D_/32, D_/32),    256, 0, stream>>>(w_proj, wT_proj, D_,   D_);
  k_transpose<<<dim3(D_/32, 4*D_/32),  256, 0, stream>>>(w_fc1,  wT_fc1,  D_,   4*D_);
  k_transpose<<<dim3(4*D_/32, D_/32),  256, 0, stream>>>(w_fc2,  wT_fc2,  4*D_, D_);

  // SiTo index pipeline
  k_xnorm <<<B_*N_,  128, 0, stream>>>(x, xn_h);
  k_window<<<B_*NW_, 128, 0, stream>>>(xn_h, noise, dsti, srci);
  k_sim   <<<B_*96,  256, 0, stream>>>(xn_h, srci, dsti, sim);
  k_rowmax<<<B_*96,  256, 0, stream>>>(sim, maxsim, best);
  k_sort  <<<B_,     512, 0, stream>>>(maxsim, best, srci, skept, sprn, pbest);
  k_gather<<<B_*T_,  128, 0, stream>>>(x, dsti, skept, xk);

  const int M = B_ * T_;  // 8192
  // DiT block
  k_ln  <<<M, 128, 0, stream>>>(xk, ln1_g, ln1_b, y_h);
  k_gemm<<<dim3(M/128, 3*D_/64), 256, 0, stream>>>(y_h, wT_qkv, b_qkv, nullptr, qkv_h, nullptr, M, 3*D_, D_, 0);
  k_repack<<<B_*H_*T_, 128, 0, stream>>>(qkv_h, qp, kp, vt);
  k_attn<<<dim3(B_*H_, T_/16), 32, 0, stream>>>(qp, kp, vt, o_h);
  k_gemm<<<dim3(M/128, D_/64),   256, 0, stream>>>(o_h, wT_proj, b_proj, xk, nullptr, x1, M, D_, D_, 2);
  k_ln  <<<M, 128, 0, stream>>>(x1, ln2_g, ln2_b, y_h);
  k_gemm<<<dim3(M/128, 4*D_/64), 256, 0, stream>>>(y_h, wT_fc1, b_fc1, nullptr, h1_h, nullptr, M, 4*D_, D_, 1);
  k_gemm<<<dim3(M/128, D_/64),   256, 0, stream>>>(h1_h, wT_fc2, b_fc2, x1, nullptr, xout, M, D_, 4*D_, 2);

  // Recover full token grid (all N positions covered exactly once)
  k_scatter<<<B_*N_, 128, 0, stream>>>(xout, dsti, skept, sprn, pbest, out);
}
